// CHIEF_12945031431004
// MI455X (gfx1250) — matrix operand order repaired and checked by reading, not compiled, not run on hardware
//
#include <hip/hip_runtime.h>

typedef __attribute__((ext_vector_type(16))) _Float16 v16h;
typedef __attribute__((ext_vector_type(8)))  _Float16 v8h;
typedef __attribute__((ext_vector_type(8)))  float    v8f;

#define NROW 50000
#define FEAT 2048
#define HID  1024
#define DATT 512
#define TXT  768

// LDS layout for stage1 (halves unless noted): strides conflict-free for b128 on 64 banks
#define SX_STRIDE 40       // 80B rows -> bank starts m*20 mod 64, distinct, 16B aligned
#define SH_STRIDE 1048     // 2096B rows -> bank starts m*12 mod 64, distinct, 16B aligned
#define SX_BUF   (32 * SX_STRIDE)            // one staging array (1280 halves)
#define SXH_OFF(b) ((b) * 2 * SX_BUF)        // buffers 0/1, hi
#define SXL_OFF(b) ((b) * 2 * SX_BUF + SX_BUF)
#define SHH_OFF (4 * SX_BUF)
#define SHL_OFF (SHH_OFF + 32 * SH_STRIDE)
#define SSC_OFF ((SHL_OFF + 32 * SH_STRIDE) * 2)   // bytes (16B aligned)
#define SMEM1_BYTES (SSC_OFF + 32 * 4)

union V16U { v16h v; v8h h[2]; };

__device__ __forceinline__ v8f v8f_zero() {
    v8f z;
#pragma unroll
    for (int i = 0; i < 8; ++i) z[i] = 0.0f;
    return z;
}

// split a float into f16 hi + f16 lo: hi + lo ~= x with ~21 mantissa bits
__device__ __forceinline__ void split_f16(float x, _Float16 &hi, _Float16 &lo) {
    _Float16 h = (_Float16)x;
    hi = h;
    lo = (_Float16)(x - (float)h);
}

__device__ __forceinline__ void store_split2(_Float16* bH, _Float16* bL, int r, int k,
                                             float v0, float v1) {
    _Float16 h0, l0, h1, l1;
    split_f16(v0, h0, l0);
    split_f16(v1, h1, l1);
    bH[r * SX_STRIDE + k] = h0;  bH[r * SX_STRIDE + k + 1] = h1;
    bL[r * SX_STRIDE + k] = l0;  bL[r * SX_STRIDE + k + 1] = l1;
}

// Load A fragment (16x32 f16) from pre-split LDS row base.
// halves j=0..15: K = 16*half + j
__device__ __forceinline__ v16h load_afrag(const _Float16* base, int half) {
    V16U u;
    u.h[0] = *(const v8h*)(base + 16 * half);
    u.h[1] = *(const v8h*)(base + 16 * half + 8);
    return u.v;
}

__device__ __forceinline__ v8f wmma3(const v16h &ahi, const v16h &alo,
                                     const v16h &bhi, const v16h &blo, v8f acc) {
    acc = __builtin_amdgcn_wmma_f32_16x16x32_f16(false, ahi, false, bhi, (short)0, acc, false, false);
    acc = __builtin_amdgcn_wmma_f32_16x16x32_f16(false, ahi, false, blo, (short)0, acc, false, false);
    acc = __builtin_amdgcn_wmma_f32_16x16x32_f16(false, alo, false, bhi, (short)0, acc, false, false);
    return acc;
}

// ---------------------------------------------------------------------------
// Pack kernel: W [K x ncols] f32 -> hi/lo f16 in B-fragment order:
// p[((k/16)*ncols + n)*16 + (k%16)] = f16(W[k][n]); one lane fragment = 32
// contiguous bytes -> one v16h load (2x global_load_b128), no hot-loop converts.
// ---------------------------------------------------------------------------
__global__ void __launch_bounds__(256)
CHIEF_pack(const float* __restrict__ W, _Float16* __restrict__ pH, _Float16* __restrict__ pL,
           int ncols)
{
    const int n = blockIdx.x * 256 + threadIdx.x;
    const int k = blockIdx.y;
    const float w = W[(size_t)k * ncols + n];
    _Float16 hi, lo;
    split_f16(w, hi, lo);
    const size_t o = ((size_t)(k >> 4) * ncols + n) * 16 + (k & 15);
    pH[o] = hi;
    pL[o] = lo;
}

// ---------------------------------------------------------------------------
// Kernel 1: per 32-row tile, fused h = relu(xW1+b1) -> s = (tanh(hWa+ba)*sig(hWb+bb))@Wc+bc
// 512 threads = 16 wave32 waves.
//  Phase A: wave w owns h-columns [w*64, +64), both 16-row groups (B reused 2x);
//           double-buffered x staging, single barrier per K-chunk, loads 2 chunks ahead.
//  Phase B: rg = w/8, cg = w%8; K outermost, A fragments shared by all 4 col tiles.
// ---------------------------------------------------------------------------
__global__ void __launch_bounds__(512)
CHIEF_stage1(const float* __restrict__ x,
             const _Float16* __restrict__ W1pH, const _Float16* __restrict__ W1pL,
             const float* __restrict__ b1,
             const _Float16* __restrict__ WapH, const _Float16* __restrict__ WapL,
             const float* __restrict__ ba,
             const _Float16* __restrict__ WbpH, const _Float16* __restrict__ WbpL,
             const float* __restrict__ bb,
             const float* __restrict__ Wc, const float* __restrict__ bc,
             float* __restrict__ Hout, float* __restrict__ scores)
{
    extern __shared__ char smemraw[];
    _Float16* const smh = (_Float16*)smemraw;
    _Float16* shH = smh + SHH_OFF;               // [32][SH_STRIDE] h tile, hi
    _Float16* shL = smh + SHL_OFF;               // [32][SH_STRIDE] h tile, lo
    float*    ssc = (float*)(smemraw + SSC_OFF); // [32] per-row score accumulators

    const int tid  = threadIdx.x;
    const int lane = tid & 31;
    const int wave = tid >> 5;
    const int half = lane >> 4;
    const int lm   = lane & 15;
    const int row0 = blockIdx.x * 32;

    // ---------------- Phase A: H = relu(X @ W1 + b1) ----------------
    v8f acc[2][4];
#pragma unroll
    for (int rg = 0; rg < 2; ++rg)
#pragma unroll
        for (int t = 0; t < 4; ++t) acc[rg][t] = v8f_zero();

    const int wcol0 = wave * 64;
    const int NKC   = FEAT / 32;

    // staging assignment: thread -> (row r, cols kk..kk+1) of the 32x32 chunk
    const int r  = tid >> 4;
    const int kk = (tid & 15) * 2;
    const int gr = row0 + r;
    const bool rvalid = (gr < NROW);
    const float* xrow = x + (size_t)(rvalid ? gr : 0) * FEAT + kk;

    // prologue: chunk0 -> buf0; chunk1 -> regs
    float p0 = 0.0f, p1 = 0.0f;
    if (rvalid) { p0 = xrow[0]; p1 = xrow[1]; }
    store_split2(smh + SXH_OFF(0), smh + SXL_OFF(0), r, kk, p0, p1);
    p0 = 0.0f; p1 = 0.0f;
    if (rvalid) { p0 = xrow[32]; p1 = xrow[33]; }
    __syncthreads();

    for (int kc = 0; kc < NKC; ++kc) {
        const int cur = kc & 1;
        // store next chunk (in regs) into the other buffer
        if (kc + 1 < NKC)
            store_split2(smh + SXH_OFF(cur ^ 1), smh + SXL_OFF(cur ^ 1), r, kk, p0, p1);
        // issue global load 2 chunks ahead (hidden behind this chunk's WMMAs)
        if (kc + 2 < NKC) {
            p0 = 0.0f; p1 = 0.0f;
            if (rvalid) { p0 = xrow[(kc + 2) * 32]; p1 = xrow[(kc + 2) * 32 + 1]; }
        }

        const _Float16* bxH = smh + SXH_OFF(cur);
        const _Float16* bxL = smh + SXL_OFF(cur);
        const v16h aH0 = load_afrag(bxH + (size_t)lm * SX_STRIDE, half);
        const v16h aL0 = load_afrag(bxL + (size_t)lm * SX_STRIDE, half);
        const v16h aH1 = load_afrag(bxH + (size_t)(16 + lm) * SX_STRIDE, half);
        const v16h aL1 = load_afrag(bxL + (size_t)(16 + lm) * SX_STRIDE, half);

        if (kc + 1 < NKC)
            __builtin_prefetch(W1pH + ((size_t)(2 * (kc + 1) + half) * HID + wcol0 + lm) * 16, 0, 1);

#pragma unroll
        for (int t = 0; t < 4; ++t) {
            const int n = wcol0 + t * 16 + lm;
            const size_t bo = ((size_t)(2 * kc + half) * HID + n) * 16;
            const v16h bhi = *(const v16h*)(W1pH + bo);
            const v16h blo = *(const v16h*)(W1pL + bo);
            acc[0][t] = wmma3(aH0, aL0, bhi, blo, acc[0][t]);   // B reused for both
            acc[1][t] = wmma3(aH1, aL1, bhi, blo, acc[1][t]);   // row groups
        }
        __syncthreads();
    }

    // epilogue: bias + relu; split-f16 h tile to LDS + f32 h to workspace
#pragma unroll
    for (int t = 0; t < 4; ++t) {
        const int n = wcol0 + t * 16 + lm;
        const float bias = b1[n];
#pragma unroll
        for (int rg = 0; rg < 2; ++rg) {
#pragma unroll
            for (int i = 0; i < 8; ++i) {
                const int m = rg * 16 + i + 8 * half;   // C layout: VGPR i -> row i (+8 upper half)
                float v = fmaxf(acc[rg][t][i] + bias, 0.0f);
                _Float16 h, l;
                split_f16(v, h, l);
                shH[(size_t)m * SH_STRIDE + n] = h;
                shL[(size_t)m * SH_STRIDE + n] = l;
                const int g2 = row0 + m;
                if (g2 < NROW) Hout[(size_t)g2 * HID + n] = v;
            }
        }
    }
    if (tid < 32) ssc[tid] = 0.0f;
    __syncthreads();

    // ---------------- Phase B: gated attention scores ----------------
    const int rg = wave >> 3;           // row group 0/1
    const int cg = wave & 7;            // 8 groups x 64 attention cols
    const _Float16* arowH = shH + (size_t)(rg * 16 + lm) * SH_STRIDE;
    const _Float16* arowL = shL + (size_t)(rg * 16 + lm) * SH_STRIDE;

    v8f accA[4], accG[4];
#pragma unroll
    for (int t = 0; t < 4; ++t) { accA[t] = v8f_zero(); accG[t] = v8f_zero(); }

    for (int kc = 0; kc < HID / 32; ++kc) {
        const int kb = kc * 32;
        const v16h ahi = load_afrag(arowH + kb, half);   // shared by all 4 tiles
        const v16h alo = load_afrag(arowL + kb, half);
#pragma unroll
        for (int t = 0; t < 4; ++t) {
            const int n = cg * 64 + t * 16 + lm;
            const size_t bo = ((size_t)(2 * kc + half) * DATT + n) * 16;
            {
                const v16h bhi = *(const v16h*)(WapH + bo);
                const v16h blo = *(const v16h*)(WapL + bo);
                accA[t] = wmma3(ahi, alo, bhi, blo, accA[t]);
            }
            {
                const v16h bhi = *(const v16h*)(WbpH + bo);
                const v16h blo = *(const v16h*)(WbpL + bo);
                accG[t] = wmma3(ahi, alo, bhi, blo, accG[t]);
            }
        }
    }

#pragma unroll
    for (int t = 0; t < 4; ++t) {
        const int n = cg * 64 + t * 16 + lm;
        const float wc  = Wc[n];
        const float bav = ba[n];
        const float bbv = bb[n];
#pragma unroll
        for (int i = 0; i < 8; ++i) {
            const float av = tanhf(accA[t][i] + bav);
            const float gv = 1.0f / (1.0f + expf(-(accG[t][i] + bbv)));
            float p = av * gv * wc;
            // reduce over the 16 tile columns: xor masks < 16 stay inside each half
#pragma unroll
            for (int off = 1; off < 16; off <<= 1)
                p += __shfl_xor(p, off, 32);
            if (lm == 0) atomicAdd(&ssc[rg * 16 + i + 8 * half], p);
        }
    }
    __syncthreads();

    if (tid < 32) {
        const int g2 = row0 + tid;
        if (g2 < NROW) scores[g2] = ssc[tid] + bc[0];
    }
}

// ---------------------------------------------------------------------------
// Kernel 2: softmax stats (max, sum of exp); also zero Mpool.
// ---------------------------------------------------------------------------
__global__ void __launch_bounds__(1024)
CHIEF_stats(const float* __restrict__ scores, float* __restrict__ stats, float* __restrict__ Mpool)
{
    __shared__ float red[1024];
    const int tid = threadIdx.x;
    Mpool[tid] = 0.0f;

    float m = -3.402823e38f;
    for (int i = tid; i < NROW; i += 1024) m = fmaxf(m, scores[i]);
    red[tid] = m;
    __syncthreads();
    for (int s = 512; s > 0; s >>= 1) {
        if (tid < s) red[tid] = fmaxf(red[tid], red[tid + s]);
        __syncthreads();
    }
    const float mx = red[0];
    __syncthreads();

    float z = 0.0f;
    for (int i = tid; i < NROW; i += 1024) z += expf(scores[i] - mx);
    red[tid] = z;
    __syncthreads();
    for (int s = 512; s > 0; s >>= 1) {
        if (tid < s) red[tid] += red[tid + s];
        __syncthreads();
    }
    if (tid == 0) { stats[0] = mx; stats[1] = red[0]; }
}

// ---------------------------------------------------------------------------
// Kernel 3: M = softmax(s) @ H   (memory-bound: 205 MB of H streamed once)
// ---------------------------------------------------------------------------
__global__ void __launch_bounds__(256)
CHIEF_pool(const float* __restrict__ H, const float* __restrict__ scores,
           const float* __restrict__ stats, float* __restrict__ Mpool)
{
    const int tid = threadIdx.x;
    const int nb  = gridDim.x;
    const int rows_per = (NROW + nb - 1) / nb;
    const int r0 = blockIdx.x * rows_per;
    const int r1 = (r0 + rows_per < NROW) ? (r0 + rows_per) : NROW;
    const float mx   = stats[0];
    const float invZ = 1.0f / stats[1];

    float a0 = 0.0f, a1 = 0.0f, a2 = 0.0f, a3 = 0.0f;
    for (int rr = r0; rr < r1; ++rr) {
        const float w = expf(scores[rr] - mx) * invZ;
        const float* hp = H + (size_t)rr * HID;
        a0 += w * hp[tid];
        a1 += w * hp[tid + 256];
        a2 += w * hp[tid + 512];
        a3 += w * hp[tid + 768];
    }
    atomicAdd(&Mpool[tid],       a0);
    atomicAdd(&Mpool[tid + 256], a1);
    atomicAdd(&Mpool[tid + 512], a2);
    atomicAdd(&Mpool[tid + 768], a3);
}

// ---------------------------------------------------------------------------
// Kernel 4: M += relu(text@Wt+bt);  logits = M @ Wcls + bcls  (tiny)
// ---------------------------------------------------------------------------
__global__ void __launch_bounds__(1024)
CHIEF_head(const float* __restrict__ text, const float* __restrict__ Wt, const float* __restrict__ bt,
           const float* __restrict__ Mpool, const float* __restrict__ Wcls, const float* __restrict__ bcls,
           float* __restrict__ out)
{
    __shared__ float red[1024];
    const int j = threadIdx.x;
    float t = bt[j];
    for (int k = 0; k < TXT; ++k) t += text[k] * Wt[(size_t)k * HID + j];
    const float Mf = Mpool[j] + fmaxf(t, 0.0f);

    for (int c = 0; c < 2; ++c) {
        red[j] = Mf * Wcls[j * 2 + c];
        __syncthreads();
        for (int s = 512; s > 0; s >>= 1) {
            if (j < s) red[j] += red[j + s];
            __syncthreads();
        }
        if (j == 0) out[c] = red[0] + bcls[c];
        __syncthreads();
    }
}

extern "C" void kernel_launch(void* const* d_in, const int* in_sizes, int n_in,
                              void* d_out, int out_size, void* d_ws, size_t ws_size,
                              hipStream_t stream) {
    (void)in_sizes; (void)n_in; (void)out_size; (void)ws_size;
    const float* x    = (const float*)d_in[0];
    const float* text = (const float*)d_in[1];
    const float* W1   = (const float*)d_in[2];
    const float* b1   = (const float*)d_in[3];
    const float* Wa   = (const float*)d_in[4];
    const float* ba   = (const float*)d_in[5];
    const float* Wb   = (const float*)d_in[6];
    const float* bb   = (const float*)d_in[7];
    const float* Wc   = (const float*)d_in[8];
    const float* bc   = (const float*)d_in[9];
    const float* Wt   = (const float*)d_in[10];
    const float* bt   = (const float*)d_in[11];
    const float* Wcls = (const float*)d_in[12];
    const float* bcls = (const float*)d_in[13];

    // workspace layout (float units; pack buffers 32B aligned)
    float* ws     = (float*)d_ws;
    float* Hout   = ws;                              // NROW*HID
    float* scores = Hout + (size_t)NROW * HID;       // NROW
    float* stats  = scores + NROW;                   // 2
    float* Mpool  = stats + 2;                       // HID
    size_t off    = (size_t)NROW * HID + NROW + 2 + HID;
    off = (off + 7) & ~(size_t)7;                    // 32B align
    _Float16* W1pH = (_Float16*)(ws + off); off += (size_t)FEAT * HID / 2;
    _Float16* W1pL = (_Float16*)(ws + off); off += (size_t)FEAT * HID / 2;
    _Float16* WapH = (_Float16*)(ws + off); off += (size_t)HID * DATT / 2;
    _Float16* WapL = (_Float16*)(ws + off); off += (size_t)HID * DATT / 2;
    _Float16* WbpH = (_Float16*)(ws + off); off += (size_t)HID * DATT / 2;
    _Float16* WbpL = (_Float16*)(ws + off);

    // one-time (per launch) weight packing: 12.6 MB of f32 weights, negligible vs GEMM
    CHIEF_pack<<<dim3(HID / 256, FEAT), 256, 0, stream>>>(W1, W1pH, W1pL, HID);
    CHIEF_pack<<<dim3(DATT / 256, HID), 256, 0, stream>>>(Wa, WapH, WapL, DATT);
    CHIEF_pack<<<dim3(DATT / 256, HID), 256, 0, stream>>>(Wb, WbpH, WbpL, DATT);

    CHIEF_stage1<<<(NROW + 31) / 32, 512, SMEM1_BYTES, stream>>>(
        x, W1pH, W1pL, b1, WapH, WapL, ba, WbpH, WbpL, bb, Wc, bc, Hout, scores);
    CHIEF_stats<<<1, 1024, 0, stream>>>(scores, stats, Mpool);
    CHIEF_pool<<<256, 256, 0, stream>>>(Hout, scores, stats, Mpool);
    CHIEF_head<<<1, 1024, 0, stream>>>(text, Wt, bt, Mpool, Wcls, bcls, (float*)d_out);
}
